// OnnxSafeAttention_18391049962053
// MI455X (gfx1250) — matrix operand order repaired and checked by reading, not compiled, not run on hardware
//
#include <hip/hip_runtime.h>
#include <hip/hip_bf16.h>

// MHA forward for MI455X (gfx1250): f16 WMMA + f32 accumulate.
// Tile data movement: TDM (tensor_load_to_lds, TENSORcnt) for uniform 2D tiles,
// GLOBAL_LOAD_ASYNC_TO_LDS_B128 (ASYNCcnt) for the flash Q tile.
// Pipeline: cvt(x,W*) -> Q/K GEMMs, V GEMM (writes V^T per head) -> flash attn
//           -> output GEMM.

typedef __attribute__((ext_vector_type(16))) _Float16 v16h;
typedef __attribute__((ext_vector_type(8)))  _Float16 v8h;
typedef __attribute__((ext_vector_type(4)))  _Float16 v4h;
typedef __attribute__((ext_vector_type(8)))  float    v8f;
typedef int v4i_ __attribute__((vector_size(16)));      // async builtin param
typedef unsigned int u32x4 __attribute__((vector_size(16)));
typedef int i32x4 __attribute__((vector_size(16)));
typedef int i32x8 __attribute__((vector_size(32)));

#define DEV __device__ __forceinline__

static constexpr int Dm  = 1024;   // embed dim
static constexpr int Sm  = 2048;   // seq len
static constexpr int Bm_ = 4;      // batch
static constexpr int Hm  = 16;     // heads
static constexpr int Hd  = 64;     // head dim
static constexpr int NR  = Bm_ * Sm;        // 8192 rows
static constexpr float ATT_SCALE = 0.125f;  // 64^-0.5

// ---------------------------------------------------------------------------
// Async global->LDS copy (ASYNCcnt)
// ---------------------------------------------------------------------------
#if __has_builtin(__builtin_amdgcn_global_load_async_to_lds_b128)
#define HAS_ASYNC_LDS 1
#else
#define HAS_ASYNC_LDS 0
#endif

DEV void cp_async16(const _Float16* g, _Float16* l) {
#if HAS_ASYNC_LDS
    __builtin_amdgcn_global_load_async_to_lds_b128(
        (__attribute__((address_space(1))) v4i_*)(g),
        (__attribute__((address_space(3))) v4i_*)(l), 0, 0);
#else
    *(v8h*)l = *(const v8h*)g;
#endif
}

DEV void async_wait0() {
#if __has_builtin(__builtin_amdgcn_s_wait_asynccnt)
    __builtin_amdgcn_s_wait_asynccnt(0);
#elif HAS_ASYNC_LDS
    asm volatile("s_wait_asynccnt 0x0" ::: "memory");
#endif
}

// ---------------------------------------------------------------------------
// Tensor Data Mover (TENSORcnt). D# per CDNA5 ISA ch.8.
// 2D tile of 16-bit elements: tile_w elems x tile_h rows, row stride in elems,
// LDS pitch via pad fields: pad every (2<<padi) DWORDs by (pada+1) DWORDs.
// This toolchain exposes the 6-arg builtin:
//   (uint32x4 g0, int32x8 g1, int32x4 g2, int32x4 g3, int32x8 g4, i32 cpol)
// ---------------------------------------------------------------------------
#if __has_builtin(__builtin_amdgcn_tensor_load_to_lds)
#define HAS_TDM 1
#else
#define HAS_TDM 0
#endif

DEV unsigned lds_off(const void* p) {
    return (unsigned)(unsigned long long)
        (__attribute__((address_space(3))) const void*)p;
}

#if HAS_TDM
DEV void tdm_load_2d(const _Float16* gsrc, unsigned ldsb,
                     int tile_w, int tile_h, long row_stride,
                     int padi, int pada) {
    const unsigned long long ga = (unsigned long long)(const void*)gsrc;
    const unsigned td0 = 1u << 20, td1 = 1u << 20;   // generous OOB dims
    u32x4 g0 = { 1u,                                  // count=1 user descriptor
                 ldsb,                                 // lds_addr
                 (unsigned)(ga & 0xffffffffu),         // global_addr[31:0]
                 (unsigned)((ga >> 32) & 0x01ffffffu) | 0x80000000u }; // type=2
    i32x8 g1 = {
        (int)((1u << 16) | (1u << 20) |               // data_size=2B, pad_enable
              ((unsigned)padi << 22) | ((unsigned)pada << 25)),
        (int)((td0 & 0xffffu) << 16),                 // [63:48]=tensor_dim0 lo16
        (int)(((td0 >> 16) & 0xffffu) |               // [79:64]=tensor_dim0 hi16
              ((td1 & 0xffffu) << 16)),               // [95:80]=tensor_dim1 lo16
        (int)(((td1 >> 16) & 0xffffu) |               // [111:96]=tensor_dim1 hi16
              ((unsigned)tile_w << 16)),              // [127:112]=tile_dim0
        (int)(unsigned)tile_h,                        // [143:128]=tile_dim1
        (int)(unsigned)row_stride,                    // dim0_stride[31:0]
        0, 0 };
    i32x4 gz4 = { 0, 0, 0, 0 };
    i32x8 gz8 = { 0, 0, 0, 0, 0, 0, 0, 0 };
    __builtin_amdgcn_tensor_load_to_lds(g0, g1, gz4, gz4, gz8, 0);
}
#endif

DEV void tensor_wait0() {
#if __has_builtin(__builtin_amdgcn_s_wait_tensorcnt)
    __builtin_amdgcn_s_wait_tensorcnt(0);
#elif HAS_TDM
    asm volatile("s_wait_tensorcnt 0x0" ::: "memory");
#endif
}

// ---------------------------------------------------------------------------
// Fragment helpers (layouts per CDNA5 ISA 7.12.2, wave32)
// ---------------------------------------------------------------------------
DEV v16h make_v16(v8h lo, v8h hi) {
    v16h r;
#pragma unroll
    for (int e = 0; e < 8; ++e) { r[e] = lo[e]; r[e + 8] = hi[e]; }
    return r;
}

DEV v16h load_a_frag(const _Float16* base, int ld, int row, int koff, int lane) {
    const int m  = row + (lane & 15);
    const int kb = koff + ((lane >> 4) & 1) * 8;
    v8h lo = *(const v8h*)(base + m * ld + kb);
    v8h hi = *(const v8h*)(base + m * ld + kb + 16);
    return make_v16(lo, hi);
}

DEV v16h load_b_frag(const _Float16* base, int ld, int col, int koff, int lane) {
    const int n  = col + (lane & 15);
    const int kb = koff + ((lane >> 4) & 1) * 8;
    v8h lo = *(const v8h*)(base + n * ld + kb);
    v8h hi = *(const v8h*)(base + n * ld + kb + 16);
    return make_v16(lo, hi);
}

DEV v8f wmma_f16(v16h a, v16h b, v8f c) {
    return __builtin_amdgcn_wmma_f32_16x16x32_f16(
        false, a, false, b, (short)0, c, false, false);
}

// ---------------------------------------------------------------------------
// f32 -> f16 elementwise convert
// ---------------------------------------------------------------------------
__global__ __launch_bounds__(256) void cvt_f32_f16(const float* __restrict__ in,
                                                   _Float16* __restrict__ out) {
    const int i = (blockIdx.x * 256 + threadIdx.x) * 4;
    float4 v = *(const float4*)(in + i);
    v4h o;
    o[0] = (_Float16)v.x; o[1] = (_Float16)v.y;
    o[2] = (_Float16)v.z; o[3] = (_Float16)v.w;
    *(v4h*)(out + i) = o;
}

// ---------------------------------------------------------------------------
// GEMM: C[M,N] = A[M,K] * W^T + bias   (W f16 [N][K] row-major -> B tile [n][k]
// is a direct 2D tile of W). block tile 128x128, BK=32, 8 waves, wave 32x64.
// Double-buffered LDS, TDM tile DMA (wave 0 issues, TENSORcnt).
// OM: 0 = f16 out; 1 = f32 out; 2 = f16 out in V^T per-head layout [b][h][d][s].
// ---------------------------------------------------------------------------
template <int OM>
__global__ __launch_bounds__(256) void gemm_wmma(const _Float16* __restrict__ A,
                                                 const _Float16* __restrict__ W,
                                                 const float* __restrict__ bias,
                                                 void* __restrict__ Cv,
                                                 int M, int N, int K) {
    constexpr int BK  = 32;
    constexpr int LDT = BK + 8;                 // 40-half pitch (16DW + 4DW pad)
    __shared__ _Float16 As[2][128 * LDT];       // [m][k]
    __shared__ _Float16 Bs[2][128 * LDT];       // [n][k]

    const int tid  = threadIdx.x;
    const int lane = tid & 31;
    const int wave = tid >> 5;
    const int wm   = wave >> 1;                 // 0..3
    const int wn   = wave & 1;                  // 0..1
    const int bm   = blockIdx.x * 128;
    const int bn   = blockIdx.y * 128;

    auto load_tiles = [&](int k0, int buf) {
#if HAS_TDM
        if (wave == 0) {
            tdm_load_2d(A + (size_t)bm * K + k0, lds_off(&As[buf][0]),
                        BK, 128, K, /*padi=*/3, /*pada=*/3);   // pitch 40 halfs
            tdm_load_2d(W + (size_t)bn * K + k0, lds_off(&Bs[buf][0]),
                        BK, 128, K, 3, 3);
        }
#else
        const int r0 = (tid * 2) >> 2, c0 = ((tid * 2) & 3) * 8;
        const int r1 = (tid * 2 + 1) >> 2, c1 = ((tid * 2 + 1) & 3) * 8;
        cp_async16(A + (size_t)(bm + r0) * K + k0 + c0, &As[buf][r0 * LDT + c0]);
        cp_async16(A + (size_t)(bm + r1) * K + k0 + c1, &As[buf][r1 * LDT + c1]);
        cp_async16(W + (size_t)(bn + r0) * K + k0 + c0, &Bs[buf][r0 * LDT + c0]);
        cp_async16(W + (size_t)(bn + r1) * K + k0 + c1, &Bs[buf][r1 * LDT + c1]);
#endif
    };
    auto wait_tiles = [&]() {
#if HAS_TDM
        if (wave == 0) tensor_wait0();
#else
        async_wait0();
#endif
        __syncthreads();
    };

    v8f acc[2][4];
#pragma unroll
    for (int i = 0; i < 2; ++i)
#pragma unroll
        for (int j = 0; j < 4; ++j) acc[i][j] = (v8f){};

    load_tiles(0, 0);
    wait_tiles();

    int cur = 0;
    for (int k0 = 0; k0 < K; k0 += BK) {
        if (k0 + BK < K) load_tiles(k0 + BK, cur ^ 1);   // DMA under compute

        v16h af0 = load_a_frag(As[cur], LDT, wm * 32, 0, lane);
        v16h af1 = load_a_frag(As[cur], LDT, wm * 32 + 16, 0, lane);
        v16h bf[4];
#pragma unroll
        for (int nt = 0; nt < 4; ++nt)
            bf[nt] = load_b_frag(Bs[cur], LDT, wn * 64 + nt * 16, 0, lane);
#pragma unroll
        for (int nt = 0; nt < 4; ++nt) {
            acc[0][nt] = wmma_f16(af0, bf[nt], acc[0][nt]);
            acc[1][nt] = wmma_f16(af1, bf[nt], acc[1][nt]);
        }

        wait_tiles();
        cur ^= 1;
    }

    const int half = (lane >> 4) & 1, l15 = lane & 15;
#pragma unroll
    for (int mt = 0; mt < 2; ++mt)
#pragma unroll
        for (int nt = 0; nt < 4; ++nt)
#pragma unroll
            for (int v = 0; v < 8; ++v) {
                const int r = bm + wm * 32 + mt * 16 + v + 8 * half;
                const int c = bn + wn * 64 + nt * 16 + l15;
                const float val = acc[mt][nt][v] + bias[c];
                if constexpr (OM == 1) {
                    ((float*)Cv)[(size_t)r * N + c] = val;
                } else if constexpr (OM == 0) {
                    ((_Float16*)Cv)[(size_t)r * N + c] = (_Float16)val;
                } else {
                    const int b = r >> 11, s = r & (Sm - 1);
                    const int h = c >> 6, d = c & (Hd - 1);
                    ((_Float16*)Cv)[((size_t)(b * Hm + h) * Hd + d) * Sm + s] =
                        (_Float16)val;
                }
            }
}

// ---------------------------------------------------------------------------
// Flash attention: one (b,h) x 64 query rows per block; 4 waves x 16 rows.
// K/V tiles: TDM double-buffered. Q tile: async-to-LDS. V arrives [d][s].
// ---------------------------------------------------------------------------
__global__ __launch_bounds__(128) void flash_attn(const _Float16* __restrict__ Qh,
                                                  const _Float16* __restrict__ Kh,
                                                  const _Float16* __restrict__ Vt,
                                                  const unsigned char* __restrict__ mask,
                                                  _Float16* __restrict__ Ah) {
    constexpr int LP = Hd + 8;            // 72-half pitch (32DW + 4DW pad)
    __shared__ _Float16 Qs[64 * LP];      // [q][d]
    __shared__ _Float16 Ps[64 * LP];      // [q][key] wave-private rows
    __shared__ _Float16 Ks[2][64 * LP];   // [key][d]
    __shared__ _Float16 Vs[2][64 * LP];   // [d][key]

    const int bh = blockIdx.x;
    const int b  = bh / Hm, h = bh % Hm;
    const int q0 = blockIdx.y * 64;
    const int tid = threadIdx.x, lane = tid & 31, wave = tid >> 5;
    const int half = (lane >> 4) & 1, l15 = lane & 15;

    const size_t base = (size_t)b * Sm * Dm + (size_t)h * Hd;
    const _Float16* Qg = Qh + base;
    const _Float16* Kg = Kh + base;
    const _Float16* Vg = Vt + (size_t)bh * Hd * Sm;   // [d][s]

    auto load_kv = [&](int j0, int buf) {
#if HAS_TDM
        if (wave == 0) {
            tdm_load_2d(Kg + (size_t)j0 * Dm, lds_off(&Ks[buf][0]),
                        Hd, 64, Dm, /*padi=*/4, /*pada=*/3);   // pitch 72 halfs
            tdm_load_2d(Vg + j0, lds_off(&Vs[buf][0]),
                        64, Hd, Sm, 4, 3);
        }
#else
#pragma unroll
        for (int c = 0; c < 4; ++c) {
            const int chunk = tid * 4 + c;
            const int row = chunk >> 3, col = (chunk & 7) * 8;
            cp_async16(Kg + (size_t)(j0 + row) * Dm + col,
                       &Ks[buf][row * LP + col]);
            cp_async16(Vg + (size_t)row * Sm + j0 + col,
                       &Vs[buf][row * LP + col]);
        }
#endif
    };
    auto wait_kv = [&]() {
#if HAS_TDM
        if (wave == 0) tensor_wait0();
#else
        async_wait0();
#endif
        __syncthreads();
    };

    // Q tile via async-to-LDS (512 chunks, 4/thread)
#pragma unroll
    for (int c = 0; c < 4; ++c) {
        const int chunk = tid * 4 + c;
        const int row = chunk >> 3, col = (chunk & 7) * 8;
        cp_async16(Qg + (size_t)(q0 + row) * Dm + col, &Qs[row * LP + col]);
    }
    load_kv(0, 0);
    async_wait0();
    wait_kv();

    float m_i[8], l_i[8];
    v8f O[4];
#pragma unroll
    for (int v = 0; v < 8; ++v) { m_i[v] = -1e30f; l_i[v] = 0.f; }
#pragma unroll
    for (int t = 0; t < 4; ++t) O[t] = (v8f){};

    int cur = 0;
    for (int j0 = 0; j0 < Sm; j0 += 64) {
        if (j0 + 64 < Sm) load_kv(j0 + 64, cur ^ 1);   // DMA under compute

        // S = Q K^T
        v8f sacc[4];
#pragma unroll
        for (int nt = 0; nt < 4; ++nt) sacc[nt] = (v8f){};
#pragma unroll
        for (int ks = 0; ks < 2; ++ks) {
            v16h a = load_a_frag(Qs, LP, wave * 16, ks * 32, lane);
            v16h bf[4];
#pragma unroll
            for (int nt = 0; nt < 4; ++nt)
                bf[nt] = load_b_frag(Ks[cur], LP, nt * 16, ks * 32, lane);
#pragma unroll
            for (int nt = 0; nt < 4; ++nt)
                sacc[nt] = wmma_f16(a, bf[nt], sacc[nt]);
        }

        // scale + key-padding mask
#pragma unroll
        for (int nt = 0; nt < 4; ++nt) {
            const bool msk = mask[(size_t)b * Sm + j0 + nt * 16 + l15] != 0;
#pragma unroll
            for (int v = 0; v < 8; ++v) {
                float s = sacc[nt][v] * ATT_SCALE;
                sacc[nt][v] = msk ? -1e30f : s;
            }
        }

        // row max -> alpha
        float alpha[8];
#pragma unroll
        for (int v = 0; v < 8; ++v) {
            float rm = sacc[0][v];
#pragma unroll
            for (int nt = 1; nt < 4; ++nt) rm = fmaxf(rm, sacc[nt][v]);
#pragma unroll
            for (int off = 8; off >= 1; off >>= 1)
                rm = fmaxf(rm, __shfl_xor(rm, off, 16));
            const float mn = fmaxf(m_i[v], rm);
            alpha[v] = __expf(m_i[v] - mn);
            m_i[v]   = mn;
        }

        // p = exp(s - m), row sums, rescale
        float rs[8];
#pragma unroll
        for (int v = 0; v < 8; ++v) rs[v] = 0.f;
#pragma unroll
        for (int nt = 0; nt < 4; ++nt)
#pragma unroll
            for (int v = 0; v < 8; ++v) {
                const float s = sacc[nt][v];
                const float p = (s > -1e29f) ? __expf(s - m_i[v]) : 0.f;
                sacc[nt][v] = p;
                rs[v] += p;
            }
#pragma unroll
        for (int v = 0; v < 8; ++v) {
#pragma unroll
            for (int off = 8; off >= 1; off >>= 1)
                rs[v] += __shfl_xor(rs[v], off, 16);
            l_i[v] = l_i[v] * alpha[v] + rs[v];
        }
#pragma unroll
        for (int t = 0; t < 4; ++t)
#pragma unroll
            for (int v = 0; v < 8; ++v) O[t][v] *= alpha[v];

        // C-layout P -> LDS (wave-private rows) -> A-layout reload
#pragma unroll
        for (int nt = 0; nt < 4; ++nt)
#pragma unroll
            for (int v = 0; v < 8; ++v) {
                const int m = wave * 16 + v + 8 * half;
                Ps[m * LP + nt * 16 + l15] = (_Float16)sacc[nt][v];
            }

        // O += P @ V
#pragma unroll
        for (int ks = 0; ks < 2; ++ks) {
            v16h a = load_a_frag(Ps, LP, wave * 16, ks * 32, lane);
            v16h bf[4];
#pragma unroll
            for (int dt = 0; dt < 4; ++dt)
                bf[dt] = load_b_frag(Vs[cur], LP, dt * 16, ks * 32, lane);
#pragma unroll
            for (int dt = 0; dt < 4; ++dt)
                O[dt] = wmma_f16(a, bf[dt], O[dt]);
        }

        wait_kv();
        cur ^= 1;
    }

    _Float16* outp = Ah + base;
#pragma unroll
    for (int dt = 0; dt < 4; ++dt)
#pragma unroll
        for (int v = 0; v < 8; ++v) {
            const int m = wave * 16 + v + 8 * half;
            const float denom = l_i[v];
            const float val = denom > 0.f ? O[dt][v] / denom : 0.f;
            outp[(size_t)(q0 + m) * Dm + dt * 16 + l15] = (_Float16)val;
        }
}

// ---------------------------------------------------------------------------
extern "C" void kernel_launch(void* const* d_in, const int* in_sizes, int n_in,
                              void* d_out, int out_size, void* d_ws, size_t ws_size,
                              hipStream_t stream) {
    const float* x  = (const float*)d_in[0];
    const unsigned char* mask = (const unsigned char*)d_in[1];  // jnp.bool_
    const float* Wq = (const float*)d_in[2];
    const float* bq = (const float*)d_in[3];
    const float* Wk = (const float*)d_in[4];
    const float* bk = (const float*)d_in[5];
    const float* Wv = (const float*)d_in[6];
    const float* bv = (const float*)d_in[7];
    const float* Wo = (const float*)d_in[8];
    const float* bo = (const float*)d_in[9];
    float* out = (float*)d_out;

    _Float16* ws = (_Float16*)d_ws;
    const size_t MAT = (size_t)NR * Dm;      // 8M halfs
    const size_t WSZ = (size_t)Dm * Dm;      // 1M halfs
    _Float16* Xh  = ws;
    _Float16* Qh  = ws + MAT;
    _Float16* Kh  = ws + 2 * MAT;
    _Float16* Vt  = ws + 3 * MAT;            // [b][h][d][s]
    _Float16* Ah  = ws + 4 * MAT;
    _Float16* Wqh = ws + 5 * MAT;
    _Float16* Wkh = Wqh + WSZ;
    _Float16* Wvh = Wkh + WSZ;
    _Float16* Woh = Wvh + WSZ;

    cvt_f32_f16<<<(int)(MAT / 1024), 256, 0, stream>>>(x, Xh);
    cvt_f32_f16<<<(int)(WSZ / 1024), 256, 0, stream>>>(Wq, Wqh);
    cvt_f32_f16<<<(int)(WSZ / 1024), 256, 0, stream>>>(Wk, Wkh);
    cvt_f32_f16<<<(int)(WSZ / 1024), 256, 0, stream>>>(Wv, Wvh);
    cvt_f32_f16<<<(int)(WSZ / 1024), 256, 0, stream>>>(Wo, Woh);

    dim3 gg(NR / 128, Dm / 128);
    gemm_wmma<0><<<gg, 256, 0, stream>>>(Xh, Wqh, bq, Qh, NR, Dm, Dm);
    gemm_wmma<0><<<gg, 256, 0, stream>>>(Xh, Wkh, bk, Kh, NR, Dm, Dm);
    gemm_wmma<2><<<gg, 256, 0, stream>>>(Xh, Wvh, bv, Vt, NR, Dm, Dm);

    dim3 fg(Bm_ * Hm, Sm / 64);
    flash_attn<<<fg, 128, 0, stream>>>(Qh, Kh, Vt, mask, Ah);

    gemm_wmma<1><<<gg, 256, 0, stream>>>(Ah, Woh, bo, out, NR, Dm, Dm);
}
